// SogouSTFT_70102456205611
// MI455X (gfx1250) — matrix operand order repaired and checked by reading, not compiled, not run on hardware
//
#include <hip/hip_runtime.h>

// ---------------------------------------------------------------------------
// MI455X (gfx1250) STFT round-trip:   y = overlap_add( frames(x) @ (fwd^T @ inv) )
//
// Strategy: fuse the two GEMMs through T = fwd^T * inv (1024x1024), emulate
// fp32 GEMM with bf16x3 split (hi*hi + hi*lo + lo*hi) on
// v_wmma_f32_16x16x32_bf16, and make overlap-add deterministic via an
// even-frame store pass + odd-frame add pass (disjoint ranges within a pass).
// ---------------------------------------------------------------------------

typedef float  v2f   __attribute__((ext_vector_type(2)));
typedef float  v8f   __attribute__((ext_vector_type(8)));
typedef __bf16 v8bf  __attribute__((ext_vector_type(8)));
typedef __bf16 v16bf __attribute__((ext_vector_type(16)));

#define FFT    1024
#define HOPSZ  512
#define CDIM   1026          // 2*CUTOFF rows of the bases
#define BATCH  32
#define NSAMP  262144        // samples per batch row
#define NFRM   511           // (NSAMP - FFT)/HOP + 1
#define XSTR   263168        // NSAMP + FFT : zero-padded per-batch stride

// ---------------------------------------------------------------------------
// Kernel 1: split x (f32) into hi/lo bf16 planes with zero-padded tail so the
// last odd-parity GEMM tile can load rows f=511 without bounds checks.
// ---------------------------------------------------------------------------
__global__ void split_kernel(const float* __restrict__ x,
                             __bf16* __restrict__ xh, __bf16* __restrict__ xl) {
  size_t idx = (size_t)blockIdx.x * blockDim.x + threadIdx.x;   // < 32*XSTR
  size_t b = idx / XSTR, n = idx % XSTR;
  float v = (n < NSAMP) ? x[b * (size_t)NSAMP + n] : 0.0f;
  __bf16 h = (__bf16)v;
  __bf16 l = (__bf16)(v - (float)h);
  xh[idx] = h;
  xl[idx] = l;
}

// ---------------------------------------------------------------------------
// Kernel 2: T = fwd^T * inv  (1024x1024, reduce over c=0..1025), computed with
// the exact-precision f32 WMMA (16x16x4).  Result stored TRANSPOSED as bf16
// hi/lo planes: Tt[k_out][k_in], so the main GEMM's B-operand is contiguous.
// One wave per 16x16 output tile; 8 waves / 256-thread block.
//
// The reduction is split into an UNGUARDED main loop (c0 = 0..1020, 256 its of
// straight-line loads + wmma — no exec-mask branching) plus a single guarded
// tail step for c = 1024/1025 (lanes 16-31 map to c=1026/1027 -> zeros).
//
// A-layout (32-bit 16x4): lanes 0-15 -> {K=c, c+1}, lanes 16-31 -> {K=c+2, c+3}
// B-layout mirrors it; C/D rows: VGPR r -> M = r (+8 for lanes 16-31).
// ---------------------------------------------------------------------------
__global__ void basisT_kernel(const float* __restrict__ fwd,
                              const float* __restrict__ inv,
                              __bf16* __restrict__ Tth, __bf16* __restrict__ Ttl) {
  const int lane = threadIdx.x & 31;
  const int wave = threadIdx.x >> 5;
  const int k2_0 = (blockIdx.x * 8 + wave) * 16;   // output column tile (k_out)
  const int k1_0 = blockIdx.y * 16;                // output row tile (k_in)
  const int M    = k1_0 + (lane & 15);
  const int N    = k2_0 + (lane & 15);
  const int chi  = (lane < 16) ? 0 : 2;

  const float* fptr = fwd + (size_t)chi * FFT + M;   // advances by 4 rows / iter
  const float* iptr = inv + (size_t)chi * FFT + N;

  v8f acc = {};
  for (int c0 = 0; c0 < 1024; c0 += 4) {           // unguarded: c+3 <= 1023+2 < 1026
    v2f a, bb;
    a[0]  = fptr[0];
    a[1]  = fptr[FFT];
    bb[0] = iptr[0];
    bb[1] = iptr[FFT];
    acc = __builtin_amdgcn_wmma_f32_16x16x4_f32(false, a, false, bb,
                                                (short)0, acc, false, false);
    fptr += 4 * FFT;
    iptr += 4 * FFT;
  }
  // Tail: c0 = 1024 -> lanes 0-15 cover c=1024,1025; lanes 16-31 would be
  // c=1026,1027 (out of range) and contribute zeros.  Single branch; EXEC is
  // all-1s again before the WMMA (ISA requirement).
  {
    v2f a = {0.0f, 0.0f}, bb = {0.0f, 0.0f};
    if (lane < 16) {
      a[0]  = fwd[(size_t)1024 * FFT + M];
      a[1]  = fwd[(size_t)1025 * FFT + M];
      bb[0] = inv[(size_t)1024 * FFT + N];
      bb[1] = inv[(size_t)1025 * FFT + N];
    }
    acc = __builtin_amdgcn_wmma_f32_16x16x4_f32(false, a, false, bb,
                                                (short)0, acc, false, false);
  }

  const int mhi = (lane < 16) ? 0 : 8;
#pragma unroll
  for (int r = 0; r < 8; ++r) {
    const int k1 = k1_0 + r + mhi;
    const float v = acc[r];
    const __bf16 h = (__bf16)v;
    const __bf16 l = (__bf16)(v - (float)h);
    Tth[(size_t)N * FFT + k1] = h;                 // transposed store
    Ttl[(size_t)N * FFT + k1] = l;
  }
}

// ---------------------------------------------------------------------------
// Kernel 3/4: frames_out = frames @ T with bf16x3 emulation, fused overlap-add.
// P = frame parity. A-tile rows are frames {P, P+2, ...} (stride-2 gather) so
// every store range within one pass is disjoint -> deterministic, atomic-free.
//   P=0: even frames exactly tile [0, 262144)  -> plain store (also inits out)
//   P=1: odd frames, disjoint among themselves -> load+add+store
//
// bf16 A-layout (16x32): lane run1 = K[off..off+7], run2 = K[off+16..off+23],
// off = 0 (lanes 0-15) or 8 (lanes 16-31); pairs per VGPR match memory order.
// bf16 B-layout (32x16): lane holds K[koff..koff+7], K[koff+16..koff+23] of one column,
// koff = 0 (lanes 0-15) or 8 (lanes 16-31) -> matches A-layout, contiguous runs in transposed T.
// ---------------------------------------------------------------------------
template <int P>
__global__ void gemm_oadd_kernel(const __bf16* __restrict__ xh,
                                 const __bf16* __restrict__ xl,
                                 const __bf16* __restrict__ Tth,
                                 const __bf16* __restrict__ Ttl,
                                 float* __restrict__ out) {
  const int lane  = threadIdx.x & 31;
  const int wave  = threadIdx.x >> 5;
  const int ktile = blockIdx.x * 8 + wave;         // 64 output-column tiles
  const int k2_0  = ktile * 16;
  const int j0    = blockIdx.y * 16;               // frame-index tile (within parity)
  const int b     = blockIdx.z;
  const int jl    = lane & 15;

  const int f = P + 2 * (j0 + jl);                 // this lane's A-matrix row
  const size_t arow = (size_t)b * XSTR + (size_t)f * HOPSZ;   // padded: f=511 reads zeros
  const int offA = (lane < 16) ? 0 : 8;
  const int offB = (lane < 16) ? 0 : 8;
  const size_t brow = (size_t)(k2_0 + jl) * FFT;

  v8f acc = {};
  for (int k0 = 0; k0 < FFT; k0 += 32) {
    const v8bf a1h = *(const v8bf*)(xh + arow + k0 + offA);
    const v8bf a2h = *(const v8bf*)(xh + arow + k0 + offA + 16);
    const v8bf a1l = *(const v8bf*)(xl + arow + k0 + offA);
    const v8bf a2l = *(const v8bf*)(xl + arow + k0 + offA + 16);
    const v8bf b1h = *(const v8bf*)(Tth + brow + k0 + offB);
    const v8bf b2h = *(const v8bf*)(Tth + brow + k0 + offB + 16);
    const v8bf b1l = *(const v8bf*)(Ttl + brow + k0 + offB);
    const v8bf b2l = *(const v8bf*)(Ttl + brow + k0 + offB + 16);

    v16bf ah, al, bh, bl;
#pragma unroll
    for (int i = 0; i < 8; ++i) {
      ah[i] = a1h[i]; ah[i + 8] = a2h[i];
      al[i] = a1l[i]; al[i + 8] = a2l[i];
      bh[i] = b1h[i]; bh[i + 8] = b2h[i];
      bl[i] = b1l[i]; bl[i + 8] = b2l[i];
    }
    // bf16x3: hi*hi + hi*lo + lo*hi ~= fp32 product
    acc = __builtin_amdgcn_wmma_f32_16x16x32_bf16(false, ah, false, bh,
                                                  (short)0, acc, false, false);
    acc = __builtin_amdgcn_wmma_f32_16x16x32_bf16(false, ah, false, bl,
                                                  (short)0, acc, false, false);
    acc = __builtin_amdgcn_wmma_f32_16x16x32_bf16(false, al, false, bh,
                                                  (short)0, acc, false, false);
  }

  const int k   = k2_0 + jl;
  const int mhi = (lane < 16) ? 0 : 8;
  const size_t obase = (size_t)b * NSAMP;
#pragma unroll
  for (int r = 0; r < 8; ++r) {
    const int fr = P + 2 * (j0 + r + mhi);
    if (P == 1 && fr > NFRM - 1) continue;         // skip padded frame 511
    const size_t n = obase + (size_t)fr * HOPSZ + k;
    if (P == 0) out[n] = acc[r];                   // even frames tile output exactly
    else        out[n] += acc[r];                  // odd frames: disjoint add
  }
}

// ---------------------------------------------------------------------------
// Host launcher.  inputs: x (32x262144 f32), forward_basis (1026x1024 f32),
// inverse_basis (1026x1024 f32).  output: y (32x262144 f32).
// Workspace: Tt_hi (2MB) | Tt_lo (2MB) | x_hi (16.06MB) | x_lo (16.06MB).
// ---------------------------------------------------------------------------
extern "C" void kernel_launch(void* const* d_in, const int* in_sizes, int n_in,
                              void* d_out, int out_size, void* d_ws, size_t ws_size,
                              hipStream_t stream) {
  const float* x   = (const float*)d_in[0];
  const float* fwd = (const float*)d_in[1];
  const float* inv = (const float*)d_in[2];
  float* out = (float*)d_out;

  __bf16* Tth = (__bf16*)d_ws;                       // 1024*1024
  __bf16* Ttl = Tth + (size_t)FFT * FFT;             // 1024*1024
  __bf16* xh  = Ttl + (size_t)FFT * FFT;             // 32*XSTR
  __bf16* xl  = xh  + (size_t)BATCH * XSTR;          // 32*XSTR

  // 1) split x into bf16 hi/lo planes (with zeroed pad tail per batch)
  split_kernel<<<(BATCH * XSTR) / 256, 256, 0, stream>>>(x, xh, xl);

  // 2) T = fwd^T * inv via exact f32 WMMA, stored transposed as bf16 hi/lo
  basisT_kernel<<<dim3(8, 64, 1), 256, 0, stream>>>(fwd, inv, Tth, Ttl);

  // 3) even-frame GEMM + store (fully covers output), 4) odd-frame GEMM + add
  gemm_oadd_kernel<0><<<dim3(8, 16, BATCH), 256, 0, stream>>>(xh, xl, Tth, Ttl, out);
  gemm_oadd_kernel<1><<<dim3(8, 16, BATCH), 256, 0, stream>>>(xh, xl, Tth, Ttl, out);
}
